// RNNFastAgentAwareness_80376017977830
// MI455X (gfx1250) — matrix operand order repaired and checked by reading, not compiled, not run on hardware
//
#include <hip/hip_runtime.h>

// ---------------- constants ----------------
#define BS   512
#define T_   64
#define NF   96
#define H_   64
#define NA_  8
#define AW_  16
#define NH_  256
#define NACT 14
#define LAT  128          // NA*AW
#define ROWS (BS * T_)    // 32768
#define B_   (BS / NA_)   // 64
#define IROWS (NA_ * B_ * NA_ * T_)  // 262144

typedef __attribute__((ext_vector_type(16))) _Float16 v16h;
typedef __attribute__((ext_vector_type(8)))  float    v8f;
typedef _Float16 h16;

#define DEVINL __device__ __forceinline__

DEVINL int lane_id() { return threadIdx.x & 31; }

// A fragment (16x32, MxK) from f16 row-major, lda in elements.
DEVINL v16h ld_a16(const h16* base, int lda) {
  int l = lane_id();
  const h16* p = base + (l & 15) * lda + ((l >> 4) << 3);
  v16h r;
#pragma unroll
  for (int i = 0; i < 8; ++i) { r[i] = p[i]; r[i + 8] = p[i + 16]; }
  return r;
}

// A fragment from f32 source (convert on the fly).
DEVINL v16h ld_a32(const float* base, int lda) {
  int l = lane_id();
  const float* p = base + (l & 15) * lda + ((l >> 4) << 3);
  v16h r;
#pragma unroll
  for (int i = 0; i < 8; ++i) { r[i] = (h16)p[i]; r[i + 8] = (h16)p[i + 16]; }
  return r;
}

// B fragment (32x16, KxN) where B[k][n] = W[n][k], W row-major [N,K] f16.
// base points at W + n0*ldw + k0.  Lane n = l&15 reads halves 8h..8h+7 and 16+8h..16+8h+7.
DEVINL v16h ld_b16(const h16* base, int ldw) {
  int l = lane_id();
  const h16* p = base + (l & 15) * ldw + ((l >> 4) << 3);
  v16h r;
#pragma unroll
  for (int i = 0; i < 8; ++i) { r[i] = p[i]; r[i + 8] = p[i + 16]; }
  return r;
}

DEVINL v8f wmma_step(v16h a, v16h b, v8f c) {
  return __builtin_amdgcn_wmma_f32_16x16x32_f16(false, a, false, b, (short)0, c,
                                                false, false);
}

DEVINL v8f acc_bias(const float* bias16) {   // bias16: 16-column base
  float bv = bias16[lane_id() & 15];
  v8f c;
#pragma unroll
  for (int r = 0; r < 8; ++r) c[r] = bv;
  return c;
}

DEVINL void st_c32(float* cbase, int ldc, v8f acc) {
  int l = lane_id(); int n = l & 15; int mb = (l >> 4) << 3;
#pragma unroll
  for (int r = 0; r < 8; ++r) cbase[(mb + r) * ldc + n] = acc[r];
}

DEVINL void st_c16(h16* cbase, int ldc, v8f acc) {
  int l = lane_id(); int n = l & 15; int mb = (l >> 4) << 3;
#pragma unroll
  for (int r = 0; r < 8; ++r) cbase[(mb + r) * ldc + n] = (h16)acc[r];
}

DEVINL float sigmoidf_(float x) { return 1.0f / (1.0f + expf(-x)); }
DEVINL float lrelu_(float x) { return x > 0.0f ? x : 0.01f * x; }

// pairwise-infer row-tile -> left/right 16x64 block bases in flat16
DEVINL void inf_bases(int rt, const h16* flat16, const h16** pl, const h16** pr) {
  int tt = rt & 3, j = (rt >> 2) & 7, b = (rt >> 5) & 63, i = rt >> 11;
  *pl = flat16 + ((size_t)((b * NA_ + i) * T_ + tt * 16)) * H_;
  *pr = flat16 + ((size_t)((b * NA_ + j) * T_ + tt * 16)) * H_;
}

// async copy of 16-byte chunks from global to LDS (ASYNCcnt path).
// ldsp must be a pointer into __shared__ memory; low 32 bits of the generic
// address are the wave's LDS byte offset (aperture mapping, ISA 10.2).
DEVINL void async_copy_b128(void* ldsp, const void* gsrc) {
  unsigned int loff = (unsigned int)(size_t)ldsp;
  unsigned long long gaddr = (unsigned long long)(size_t)gsrc;
  asm volatile("global_load_async_to_lds_b128 %0, %1, off"
               :: "v"(loff), "v"(gaddr) : "memory");
}
DEVINL void async_wait0() {
  asm volatile("s_wait_asynccnt 0x0" ::: "memory");
}

// ---------------- utility kernels ----------------
__global__ void k_cvt(const float* __restrict__ s, h16* __restrict__ d, int n) {
  int i = blockIdx.x * 256 + threadIdx.x;
  if (i < n) d[i] = (h16)s[i];
}
__global__ void k_zero32(float* d, int n) {
  int i = blockIdx.x * 256 + threadIdx.x;
  if (i < n) d[i] = 0.0f;
}
__global__ void k_zero16(h16* d, int n) {
  int i = blockIdx.x * 256 + threadIdx.x;
  if (i < n) d[i] = (h16)0.0f;
}

// ---------------- fc1: relu(inputs @ fc1_w^T + b) -> x_f16 [ROWS,64] ----------
__global__ void k_fc1(const float* __restrict__ in, const h16* __restrict__ w,
                      const float* __restrict__ b, h16* __restrict__ xo) {
  int task = blockIdx.x * 8 + (threadIdx.x >> 5);  // 8192 tasks
  int rt = task >> 2, ct = task & 3;
  const float* abase = in + (size_t)rt * 16 * NF;
  __builtin_prefetch(abase + 16 * NF, 0, 1);
  v8f acc = acc_bias(b + ct * 16);
#pragma unroll
  for (int ks = 0; ks < 3; ++ks)
    acc = wmma_step(ld_a32(abase + ks * 32, NF),
                    ld_b16(w + ct * 16 * NF + ks * 32, NF), acc);
#pragma unroll
  for (int r = 0; r < 8; ++r) acc[r] = fmaxf(acc[r], 0.0f);
  st_c16(xo + (size_t)rt * 16 * H_ + ct * 16, H_, acc);
}

// ---------------- GRU: persistent over batch-row tiles -----------------------
__global__ void k_gru(const h16* __restrict__ x16, const float* __restrict__ h0,
                      const h16* __restrict__ wih, const h16* __restrict__ whh,
                      const float* __restrict__ bih, const float* __restrict__ bhh,
                      float* __restrict__ flat_out, h16* __restrict__ flat16,
                      float* __restrict__ hT_out) {
  __shared__ float g[2][16][192];
  __shared__ float g2[2][16][64];
  __shared__ float hf[2][16][64];
  __shared__ h16   hh[2][16][64];
  int w = threadIdx.x >> 5;
  int l = threadIdx.x & 31;
  int b0 = (blockIdx.x * 2 + w) * 16;
  int m = l >> 1, cb = (l & 1) * 32;
#pragma unroll 4
  for (int c = cb; c < cb + 32; ++c) {
    float v = h0[(b0 + m) * H_ + c];
    hf[w][m][c] = v; hh[w][m][c] = (h16)v;
  }
  __syncthreads();
  for (int t = 0; t < T_; ++t) {
    const h16* xbase = x16 + ((size_t)b0 * T_ + t) * H_;   // lda = T_*H_
    __builtin_prefetch(xbase + H_, 0, 1);
    // r,z gates: gi+gh combined (cols 0..127)
    for (int ct = 0; ct < 8; ++ct) {
      int c0 = ct * 16;
      float bv = bih[c0 + (l & 15)] + bhh[c0 + (l & 15)];
      v8f acc;
#pragma unroll
      for (int r = 0; r < 8; ++r) acc[r] = bv;
#pragma unroll
      for (int ks = 0; ks < 2; ++ks) {
        acc = wmma_step(ld_a16(xbase + ks * 32, T_ * H_),
                        ld_b16(wih + c0 * H_ + ks * 32, H_), acc);
        acc = wmma_step(ld_a16(&hh[w][0][ks * 32], H_),
                        ld_b16(whh + c0 * H_ + ks * 32, H_), acc);
      }
      st_c32(&g[w][0][c0], 192, acc);
    }
    // n gate: gi and gh kept separate (cols 128..191)
    for (int ct = 0; ct < 4; ++ct) {
      int c0 = 128 + ct * 16;
      v8f ai = acc_bias(bih + c0);
      v8f ah = acc_bias(bhh + c0);
#pragma unroll
      for (int ks = 0; ks < 2; ++ks) {
        ai = wmma_step(ld_a16(xbase + ks * 32, T_ * H_),
                       ld_b16(wih + c0 * H_ + ks * 32, H_), ai);
        ah = wmma_step(ld_a16(&hh[w][0][ks * 32], H_),
                       ld_b16(whh + c0 * H_ + ks * 32, H_), ah);
      }
      st_c32(&g[w][0][c0], 192, ai);
      st_c32(&g2[w][0][ct * 16], 64, ah);
    }
    __syncthreads();
#pragma unroll 4
    for (int c = cb; c < cb + 32; ++c) {
      float rg = sigmoidf_(g[w][m][c]);
      float zg = sigmoidf_(g[w][m][64 + c]);
      float ng = tanhf(g[w][m][128 + c] + rg * g2[w][m][c]);
      float hn = (1.0f - zg) * ng + zg * hf[w][m][c];
      hf[w][m][c] = hn; hh[w][m][c] = (h16)hn;
      size_t fo = ((size_t)(b0 + m) * T_ + t) * H_ + c;
      flat_out[fo] = hn; flat16[fo] = (h16)hn;
    }
    __syncthreads();
  }
#pragma unroll 4
  for (int c = cb; c < cb + 32; ++c)
    hT_out[(b0 + m) * H_ + c] = hf[w][m][c];
}

// ---------------- BN column stats (GEMM + atomic col sums) -------------------
DEVINL void stat_reduce(v8f acc, int ct, float* sum, float* sq) {
  float s = 0.0f, s2 = 0.0f;
#pragma unroll
  for (int r = 0; r < 8; ++r) { s += acc[r]; s2 += acc[r] * acc[r]; }
  s  += __shfl_xor(s, 16, 32);
  s2 += __shfl_xor(s2, 16, 32);
  int l = lane_id();
  if (l < 16) {
    atomicAdd(&sum[ct * 16 + l], s);
    atomicAdd(&sq[ct * 16 + l], s2);
  }
}

__global__ void k_stats_aw(const h16* __restrict__ flat16, const h16* __restrict__ w,
                           const float* __restrict__ b, float* sum, float* sq) {
  int task = blockIdx.x * 8 + (threadIdx.x >> 5);   // 32768
  int rt = task >> 4, ct = task & 15;
  const h16* abase = flat16 + (size_t)rt * 16 * H_;
  v8f acc = acc_bias(b + ct * 16);
#pragma unroll
  for (int ks = 0; ks < 2; ++ks)
    acc = wmma_step(ld_a16(abase + ks * 32, H_),
                    ld_b16(w + ct * 16 * H_ + ks * 32, H_), acc);
  stat_reduce(acc, ct, sum, sq);
}

__global__ void k_stats_inf(const h16* __restrict__ flat16, const h16* __restrict__ w,
                            const float* __restrict__ b, float* sum, float* sq) {
  int task = blockIdx.x * 8 + (threadIdx.x >> 5);   // 262144
  int rt = task >> 4, ct = task & 15;
  const h16 *pl, *pr; inf_bases(rt, flat16, &pl, &pr);
  v8f acc = acc_bias(b + ct * 16);
#pragma unroll
  for (int ks = 0; ks < 2; ++ks) {
    acc = wmma_step(ld_a16(pl + ks * 32, H_),
                    ld_b16(w + ct * 16 * 128 + ks * 32, 128), acc);
    acc = wmma_step(ld_a16(pr + ks * 32, H_),
                    ld_b16(w + ct * 16 * 128 + 64 + ks * 32, 128), acc);
  }
  stat_reduce(acc, ct, sum, sq);
}

__global__ void k_finalize(const float* sum, const float* sq, float inv_n,
                           float* mean, float* rstd) {
  int i = threadIdx.x;   // 256
  float m = sum[i] * inv_n;
  float v = sq[i] * inv_n - m * m;
  mean[i] = m;
  rstd[i] = rsqrtf(v + 1e-5f);
}

// ------ awareness pass: (GEMM -> BN -> lrelu)->LDS f16 -> aw2 GEMM -> ap -----
__global__ void k_aw_pass(const h16* __restrict__ flat16, const h16* __restrict__ w1,
                          const float* __restrict__ b1, const float* __restrict__ mean,
                          const float* __restrict__ rstd, const float* __restrict__ bng,
                          const float* __restrict__ bnb, const h16* __restrict__ w2,
                          const float* __restrict__ b2, float* __restrict__ ap) {
  __shared__ h16 mid[4][16][256];
  int w = threadIdx.x >> 5;
  int rt = blockIdx.x * 4 + w;
  const h16* abase = flat16 + (size_t)rt * 16 * H_;
  int l = threadIdx.x & 31, n = l & 15, mb = (l >> 4) << 3;
  for (int ct = 0; ct < 16; ++ct) {
    int c = ct * 16 + n;
    v8f acc = acc_bias(b1 + ct * 16);
#pragma unroll
    for (int ks = 0; ks < 2; ++ks)
      acc = wmma_step(ld_a16(abase + ks * 32, H_),
                      ld_b16(w1 + ct * 16 * H_ + ks * 32, H_), acc);
    float mu = mean[c], rs = rstd[c], gg = bng[c], bb = bnb[c];
#pragma unroll
    for (int r = 0; r < 8; ++r)
      mid[w][mb + r][c] = (h16)lrelu_((acc[r] - mu) * rs * gg + bb);
  }
  __syncthreads();
  for (int ot = 0; ot < 16; ++ot) {
    v8f acc = acc_bias(b2 + ot * 16);
#pragma unroll
    for (int ks = 0; ks < 8; ++ks)
      acc = wmma_step(ld_a16(&mid[w][0][ks * 32], 256),
                      ld_b16(w2 + ot * 16 * 256 + ks * 32, 256), acc);
    st_c32(ap + (size_t)rt * 16 * 256 + ot * 16, 256, acc);
  }
}

// ------ pairwise infer pass: gathered GEMM -> BN -> lrelu -> inf2 -> inf_out -
// inf2 weights (32x256 f16 = 16KB) staged once per block into LDS via the
// async global->LDS path (ASYNCcnt), overlapped with phase-1 WMMA work.
__global__ void k_inf_pass(const h16* __restrict__ flat16, const h16* __restrict__ w1,
                           const float* __restrict__ b1, const float* __restrict__ mean,
                           const float* __restrict__ rstd, const float* __restrict__ bng,
                           const float* __restrict__ bnb, const h16* __restrict__ w2,
                           const float* __restrict__ b2, float* __restrict__ outp) {
  __shared__ h16 mid[4][16][256];
  __shared__ h16 w2l[32 * 256];
  // kick off async weight staging: 16KB = 1024 x b128, 128 threads x 8 chunks
#pragma unroll
  for (int it = 0; it < 8; ++it) {
    int chunk = it * 128 + (int)threadIdx.x;           // 16-byte chunk index
    async_copy_b128(((char*)w2l) + chunk * 16, ((const char*)w2) + chunk * 16);
  }
  int w = threadIdx.x >> 5;
  int rt = blockIdx.x * 4 + w;
  const h16 *pl, *pr; inf_bases(rt, flat16, &pl, &pr);
  int l = threadIdx.x & 31, n = l & 15, mb = (l >> 4) << 3;
  for (int ct = 0; ct < 16; ++ct) {
    int c = ct * 16 + n;
    v8f acc = acc_bias(b1 + ct * 16);
#pragma unroll
    for (int ks = 0; ks < 2; ++ks) {
      acc = wmma_step(ld_a16(pl + ks * 32, H_),
                      ld_b16(w1 + ct * 16 * 128 + ks * 32, 128), acc);
      acc = wmma_step(ld_a16(pr + ks * 32, H_),
                      ld_b16(w1 + ct * 16 * 128 + 64 + ks * 32, 128), acc);
    }
    float mu = mean[c], rs = rstd[c], gg = bng[c], bb = bnb[c];
#pragma unroll
    for (int r = 0; r < 8; ++r)
      mid[w][mb + r][c] = (h16)lrelu_((acc[r] - mu) * rs * gg + bb);
  }
  async_wait0();       // weights resident in LDS
  __syncthreads();     // mid + w2l visible block-wide
  for (int ot = 0; ot < 2; ++ot) {
    v8f acc = acc_bias(b2 + ot * 16);
#pragma unroll
    for (int ks = 0; ks < 8; ++ks)
      acc = wmma_step(ld_a16(&mid[w][0][ks * 32], 256),
                      ld_b16(&w2l[ot * 16 * 256 + ks * 32], 256), acc);
    st_c32(outp + (size_t)rt * 16 * 32 + ot * 16, 32, acc);
  }
}

// ---------------- rsample: awareness = mu + sqrt(var)*noise  -> f16 ----------
__global__ void k_rsample(const float* __restrict__ ap, const float* __restrict__ noise,
                          h16* __restrict__ aw16) {
  size_t i = (size_t)blockIdx.x * 256 + threadIdx.x;   // ROWS*128
  size_t row = i >> 7; int l = (int)(i & 127);
  float mu = ap[row * 256 + l];
  float vr = fmaxf(expf(ap[row * 256 + 128 + l]), 0.002f);
  aw16[i] = (h16)(mu + sqrtf(vr) * noise[i]);
}

// ---------------- MHA over 8 agents, 4 heads; lane = (head, agent) ----------
__global__ void k_mha(const h16* __restrict__ aw16, const float* __restrict__ wq,
                      const float* __restrict__ bq, const float* __restrict__ wk,
                      const float* __restrict__ bk, const float* __restrict__ wv,
                      const float* __restrict__ bv_, const float* __restrict__ wo,
                      const float* __restrict__ bo, h16* __restrict__ ao16) {
  __shared__ float olds[8][8][128];
  int w = threadIdx.x >> 5, l = threadIdx.x & 31;
  size_t n = (size_t)blockIdx.x * 8 + w;
  int h = l >> 3, aq = l & 7;
  float xr[16];
#pragma unroll
  for (int f = 0; f < 16; ++f) xr[f] = (float)aw16[n * 128 + aq * 16 + f];
  float q[32], k[32], v[32];
#pragma unroll
  for (int d = 0; d < 32; ++d) {
    int row = h * 32 + d;
    float s0 = bq[row], s1 = bk[row], s2 = bv_[row];
#pragma unroll
    for (int f = 0; f < 16; ++f) {
      s0 += xr[f] * wq[row * 16 + f];
      s1 += xr[f] * wk[row * 16 + f];
      s2 += xr[f] * wv[row * 16 + f];
    }
    q[d] = s0; k[d] = s1; v[d] = s2;
  }
  const float inv = 0.17677669529663687f;  // 1/sqrt(32)
  float sc[8];
#pragma unroll
  for (int j = 0; j < 8; ++j) {
    float s = 0.0f;
#pragma unroll
    for (int d = 0; d < 32; ++d) s += q[d] * __shfl(k[d], (h << 3) + j, 32);
    sc[j] = s * inv;
  }
  float mx = sc[0];
#pragma unroll
  for (int j = 1; j < 8; ++j) mx = fmaxf(mx, sc[j]);
  float se = 0.0f;
#pragma unroll
  for (int j = 0; j < 8; ++j) { sc[j] = expf(sc[j] - mx); se += sc[j]; }
  float rs = 1.0f / se;
#pragma unroll
  for (int j = 0; j < 8; ++j) sc[j] *= rs;
#pragma unroll
  for (int d = 0; d < 32; ++d) {
    float o = 0.0f;
#pragma unroll
    for (int j = 0; j < 8; ++j) o += sc[j] * __shfl(v[d], (h << 3) + j, 32);
    olds[w][aq][h * 32 + d] = o;
  }
  __syncthreads();
  int a = l >> 2, cb = (l & 3) * 4;
  for (int c = cb; c < cb + 4; ++c) {
    float s = bo[c];
#pragma unroll 8
    for (int kk = 0; kk < 128; ++kk) s += olds[w][a][kk] * wo[c * 128 + kk];
    ao16[n * 128 + a * 16 + c] = (h16)s;
  }
}

// ---------------- fc2: q = [flat | ao] @ fc2_w^T + b  (N padded 14->16) ------
__global__ void k_fc2(const h16* __restrict__ flat16, const h16* __restrict__ ao16,
                      const h16* __restrict__ wp, const float* __restrict__ b,
                      float* __restrict__ qout) {
  int rt = blockIdx.x * 8 + (threadIdx.x >> 5);   // 2048 row tiles
  int l = threadIdx.x & 31, n = l & 15, mb = (l >> 4) << 3;
  float bv = (n < NACT) ? b[n] : 0.0f;
  v8f acc;
#pragma unroll
  for (int r = 0; r < 8; ++r) acc[r] = bv;
#pragma unroll
  for (int ks = 0; ks < 2; ++ks)
    acc = wmma_step(ld_a16(flat16 + (size_t)rt * 16 * H_ + ks * 32, H_),
                    ld_b16(wp + ks * 32, 192), acc);
#pragma unroll
  for (int ks = 0; ks < 4; ++ks)
    acc = wmma_step(ld_a16(ao16 + (size_t)rt * 16 * 128 + ks * 32, 128),
                    ld_b16(wp + 64 + ks * 32, 192), acc);
  if (n < NACT) {
#pragma unroll
    for (int r = 0; r < 8; ++r)
      qout[((size_t)rt * 16 + mb + r) * NACT + n] = acc[r];
  }
}

// ---------------- KLD reduction: one block per (b, agent) --------------------
__global__ void k_kld(const float* __restrict__ ap, const float* __restrict__ infout,
                      float* __restrict__ kout) {
  __shared__ float red[256];
  int idx = blockIdx.x;   // b*8 + a
  float s = 0.0f;
  for (int e = threadIdx.x; e < T_ * LAT; e += 256) {
    int t = e >> 7, l = e & 127;
    size_t row = (size_t)idx * T_ + t;
    float mu1 = ap[row * 256 + l];
    float v1 = fmaxf(expf(ap[row * 256 + 128 + l]), 0.002f);
    size_t L = row * 128 + l;
    size_t r2 = L >> 4; int c = (int)(L & 15);
    float mu2 = infout[r2 * 32 + c];
    float v2 = fmaxf(expf(infout[r2 * 32 + 16 + c]), 0.002f);
    float dm = mu1 - mu2;
    s += 0.5f * logf(v2 / v1) + 0.5f * (v1 + dm * dm) / v2 - 0.5f;
  }
  red[threadIdx.x] = s;
  __syncthreads();
  for (int st = 128; st > 0; st >>= 1) {
    if (threadIdx.x < st) red[threadIdx.x] += red[threadIdx.x + st];
    __syncthreads();
  }
  if (threadIdx.x == 0) kout[idx] = red[0] * (1.0f / (T_ * LAT));
}

// ---------------- host launcher ----------------
extern "C" void kernel_launch(void* const* d_in, const int* in_sizes, int n_in,
                              void* d_out, int out_size, void* d_ws, size_t ws_size,
                              hipStream_t stream) {
  (void)in_sizes; (void)n_in; (void)out_size; (void)ws_size;
  const float* inputs  = (const float*)d_in[0];
  const float* h0      = (const float*)d_in[1];
  const float* noise   = (const float*)d_in[2];
  const float* fc1_w   = (const float*)d_in[3];
  const float* fc1_b   = (const float*)d_in[4];
  const float* gru_wih = (const float*)d_in[5];
  const float* gru_whh = (const float*)d_in[6];
  const float* gru_bih = (const float*)d_in[7];
  const float* gru_bhh = (const float*)d_in[8];
  const float* aw1_w   = (const float*)d_in[9];
  const float* aw1_b   = (const float*)d_in[10];
  const float* aw_bn_g = (const float*)d_in[11];
  const float* aw_bn_b = (const float*)d_in[12];
  const float* aw2_w   = (const float*)d_in[13];
  const float* aw2_b   = (const float*)d_in[14];
  const float* inf1_w  = (const float*)d_in[15];
  const float* inf1_b  = (const float*)d_in[16];
  const float* inf_bn_g= (const float*)d_in[17];
  const float* inf_bn_b= (const float*)d_in[18];
  const float* inf2_w  = (const float*)d_in[19];
  const float* inf2_b  = (const float*)d_in[20];
  const float* wq      = (const float*)d_in[21];
  const float* bq      = (const float*)d_in[22];
  const float* wk      = (const float*)d_in[23];
  const float* bk      = (const float*)d_in[24];
  const float* wv      = (const float*)d_in[25];
  const float* bv      = (const float*)d_in[26];
  const float* wo      = (const float*)d_in[27];
  const float* bo      = (const float*)d_in[28];
  const float* fc2_w   = (const float*)d_in[29];
  const float* fc2_b   = (const float*)d_in[30];

  float* out_q    = (float*)d_out;                       // 32768*14
  float* out_h    = out_q + (size_t)ROWS * NACT;         // 512*64
  float* out_flat = out_h + (size_t)BS * H_;             // 32768*64
  float* out_kld  = out_flat + (size_t)ROWS * H_;        // 512

  char* base = (char*)d_ws;
  size_t off = 0;
  auto alloc = [&](size_t bytes) -> void* {
    off = (off + 255) & ~(size_t)255;
    void* p = base + off;
    off += bytes;
    return p;
  };
  h16* w_fc1  = (h16*)alloc((size_t)H_ * NF * 2);
  h16* w_wih  = (h16*)alloc((size_t)192 * H_ * 2);
  h16* w_whh  = (h16*)alloc((size_t)192 * H_ * 2);
  h16* w_aw1  = (h16*)alloc((size_t)NH_ * H_ * 2);
  h16* w_aw2  = (h16*)alloc((size_t)256 * NH_ * 2);
  h16* w_inf1 = (h16*)alloc((size_t)NH_ * 128 * 2);
  h16* w_inf2 = (h16*)alloc((size_t)32 * NH_ * 2);
  h16* w_fc2p = (h16*)alloc((size_t)16 * 192 * 2);       // padded 14->16 rows
  h16* x16    = (h16*)alloc((size_t)ROWS * H_ * 2);
  h16* flat16 = (h16*)alloc((size_t)ROWS * H_ * 2);
  h16* aw16   = (h16*)alloc((size_t)ROWS * LAT * 2);
  h16* ao16   = (h16*)alloc((size_t)ROWS * LAT * 2);
  float* ap     = (float*)alloc((size_t)ROWS * 256 * 4);
  float* infout = (float*)alloc((size_t)IROWS * 32 * 4);
  float* stats  = (float*)alloc((size_t)2048 * 4);
  float* aw_sum = stats, *aw_sq = stats + 256;
  float* inf_sum = stats + 512, *inf_sq = stats + 768;
  float* aw_mean = stats + 1024, *aw_rstd = stats + 1280;
  float* inf_mean = stats + 1536, *inf_rstd = stats + 1792;

  // stats must be zeroed every call (atomics accumulate)
  k_zero32<<<4, 256, 0, stream>>>(stats, 1024);
  // weight conversions
  k_cvt<<<(6144 + 255) / 256, 256, 0, stream>>>(fc1_w, w_fc1, 6144);
  k_cvt<<<(12288 + 255) / 256, 256, 0, stream>>>(gru_wih, w_wih, 12288);
  k_cvt<<<(12288 + 255) / 256, 256, 0, stream>>>(gru_whh, w_whh, 12288);
  k_cvt<<<(16384 + 255) / 256, 256, 0, stream>>>(aw1_w, w_aw1, 16384);
  k_cvt<<<(65536 + 255) / 256, 256, 0, stream>>>(aw2_w, w_aw2, 65536);
  k_cvt<<<(32768 + 255) / 256, 256, 0, stream>>>(inf1_w, w_inf1, 32768);
  k_cvt<<<(8192 + 255) / 256, 256, 0, stream>>>(inf2_w, w_inf2, 8192);
  k_zero16<<<(3072 + 255) / 256, 256, 0, stream>>>(w_fc2p, 3072);
  k_cvt<<<(2688 + 255) / 256, 256, 0, stream>>>(fc2_w, w_fc2p, 2688);

  // fc1 -> x16
  k_fc1<<<1024, 256, 0, stream>>>(inputs, w_fc1, fc1_b, x16);
  // GRU -> flat (f32, to d_out), flat16, hT
  k_gru<<<16, 64, 0, stream>>>(x16, h0, w_wih, w_whh, gru_bih, gru_bhh,
                               out_flat, flat16, out_h);
  // BN batch stats
  k_stats_aw<<<4096, 256, 0, stream>>>(flat16, w_aw1, aw1_b, aw_sum, aw_sq);
  k_stats_inf<<<32768, 256, 0, stream>>>(flat16, w_inf1, inf1_b, inf_sum, inf_sq);
  k_finalize<<<1, 256, 0, stream>>>(aw_sum, aw_sq, 1.0f / (float)ROWS, aw_mean, aw_rstd);
  k_finalize<<<1, 256, 0, stream>>>(inf_sum, inf_sq, 1.0f / (float)IROWS, inf_mean, inf_rstd);
  // awareness MLP -> ap
  k_aw_pass<<<512, 128, 0, stream>>>(flat16, w_aw1, aw1_b, aw_mean, aw_rstd,
                                     aw_bn_g, aw_bn_b, w_aw2, aw2_b, ap);
  // rsample -> aw16
  k_rsample<<<16384, 256, 0, stream>>>(ap, noise, aw16);
  // pairwise infer MLP -> infout  (async LDS weight staging inside)
  k_inf_pass<<<4096, 128, 0, stream>>>(flat16, w_inf1, inf1_b, inf_mean, inf_rstd,
                                       inf_bn_g, inf_bn_b, w_inf2, inf2_b, infout);
  // MHA -> ao16
  k_mha<<<4096, 256, 0, stream>>>(aw16, wq, bq, wk, bk, wv, bv, wo, bo, ao16);
  // fc2 -> q
  k_fc2<<<256, 256, 0, stream>>>(flat16, ao16, w_fc2p, fc2_b, out_q);
  // KLD -> kld
  k_kld<<<512, 256, 0, stream>>>(ap, infout, out_kld);
}
